// GCN_MME_4784593567773
// MI455X (gfx1250) — compile-verified, matrix operand order checked by reading, not run on hardware
//
#include <hip/hip_runtime.h>
#include <hip/hip_bf16.h>

typedef unsigned short ushort_t;
typedef __attribute__((ext_vector_type(16))) __bf16 v16bf;
typedef __attribute__((ext_vector_type(16), aligned(16))) __bf16 v16bf_a16; // 16B-aligned view
typedef __attribute__((ext_vector_type(8)))  float  v8f;

#define BM 128
#define BN 128
#define BK 32
#define LDSS 40   // ushorts per LDS row (80 B: conflict-free 4-bank spacing for b128 frags)

#if defined(__AMDGCN__) && __has_builtin(__builtin_amdgcn_global_load_async_to_lds_b128)
#define ASYNC_LDS 1
typedef __attribute__((ext_vector_type(4))) int v4i;
typedef __attribute__((address_space(1))) v4i v4i_g;   // global (AS1) int4
typedef __attribute__((address_space(3))) v4i v4i_l;   // LDS    (AS3) int4
#endif

__device__ __forceinline__ void wait_async_zero() {
#ifdef ASYNC_LDS
#if __has_builtin(__builtin_amdgcn_s_wait_asynccnt)
    __builtin_amdgcn_s_wait_asynccnt(0);
#else
    asm volatile("s_wait_asynccnt 0x0" ::: "memory");
#endif
#endif
}

__device__ __forceinline__ ushort_t f2bf(float f) {
    union { __bf16 b; ushort_t u; } c;
    c.b = (__bf16)f;                       // hardware RTNE f32->bf16
    return c.u;
}

// ---------------------------------------------------------------------------
// Fused WMMA GEMM:  Out[m] = epilogue( A[m] (Nrows x Kact, lda) @ Bt[m]^T )
//   Bt is pre-transposed bf16 [Cpad x Kpad] (zero padded) so B-columns are
//   contiguous. 256 threads = 8 waves; workgroup tile 128x128; each wave
//   computes 32x64 via 2x4 16x16x32 bf16 WMMA fragments. LDS double-buffered;
//   B (and bf16 A) tiles staged via async global->LDS copies when available.
// MODE 0: out_bf16 = acc*alpha[c] + beta[c]          (BatchNorm folded)
// MODE 1: atomicAdd(out_f32, 0.5f*(acc + beta[c]))   (decoder mean over m)
// ---------------------------------------------------------------------------
template<int MODE, bool A_F32>
__global__ __launch_bounds__(256)
void gemm_wmma(const void* __restrict__ Av, long long strideAm, int lda,
               int Kact, int Kpad,
               const ushort_t* __restrict__ Bt, long long strideBm, int ldb,
               void* __restrict__ Out, long long strideOm, int ldc,
               const float* __restrict__ alpha, const float* __restrict__ beta,
               int ab_stride, int Nrows, int Ncols)
{
    __shared__ ushort_t lA[2][BM * LDSS];
    __shared__ ushort_t lB[2][BN * LDSS];

    const int m       = blockIdx.z;
    const int rowBase = blockIdx.x * BM;
    const int colBase = blockIdx.y * BN;
    const int tid  = threadIdx.x;
    const int lane = tid & 31;
    const int wave = tid >> 5;
    const int wr = (wave >> 1) * 32;   // wave row offset inside tile (0..96)
    const int wc = (wave & 1) * 64;    // wave col offset inside tile (0/64)

    const ushort_t* Bm = Bt + (long long)m * strideBm;

    // cooperative-load coordinates: 256 threads cover 128 rows x 2 k-halves
    const int ldRow = tid >> 1;          // 0..127
    const int ldKh  = (tid & 1) * 16;    // 0 or 16

    const int  aRow   = rowBase + ldRow;
    const bool aRowOk = aRow < Nrows;

    const float*    Agf = (const float*)Av    + (A_F32 ? ((long long)m * strideAm + (long long)aRow * lda + ldKh) : 0);
    const ushort_t* Agh = (const ushort_t*)Av + (A_F32 ? 0 : ((long long)m * strideAm + (long long)aRow * lda + ldKh));
    const ushort_t* Bg  = Bm + (long long)(colBase + ldRow) * ldb + ldKh;

    auto stage = [&](int k0, int buf) {
        ushort_t* la = &lA[buf][ldRow * LDSS + ldKh];
        // ---- A tile ----
        if (!aRowOk) {
            v16bf_a16 z = {};
            *(v16bf_a16*)la = z;                         // dead rows: zero, no loads
        } else if (A_F32) {
            __bf16 tmp[16] __attribute__((aligned(32)));
            const float* A = Agf + k0;
            if ((k0 + BK) <= Kact) {
                __builtin_prefetch(A + 2 * BK, 0, 1);
                #pragma unroll
                for (int q = 0; q < 4; ++q) {
                    float4 f = *(const float4*)(A + q * 4);
                    tmp[q*4+0] = (__bf16)f.x; tmp[q*4+1] = (__bf16)f.y;
                    tmp[q*4+2] = (__bf16)f.z; tmp[q*4+3] = (__bf16)f.w;
                }
            } else {
                #pragma unroll
                for (int i = 0; i < 16; ++i) {           // K tail (zero-fills k >= Kact)
                    int kg = k0 + ldKh + i;
                    tmp[i] = (kg < Kact) ? (__bf16)A[i] : (__bf16)0.f;
                }
            }
            *(v16bf_a16*)la = *(v16bf_a16*)tmp;
        } else {
            const ushort_t* A = Agh + k0;
            if ((k0 + BK) <= Kact) {
#ifdef ASYNC_LDS
                __builtin_amdgcn_global_load_async_to_lds_b128(
                    (v4i_g*)(void*)A, (v4i_l*)(void*)la, 0, 0);
                __builtin_amdgcn_global_load_async_to_lds_b128(
                    (v4i_g*)(void*)A, (v4i_l*)(void*)la, 16, 0);
#else
                *(v16bf_a16*)la = *(const v16bf_a16*)A;
#endif
            } else {
                ushort_t tu[16] __attribute__((aligned(32)));
                #pragma unroll
                for (int i = 0; i < 16; ++i) {
                    int kg = k0 + ldKh + i;
                    tu[i] = (kg < Kact) ? A[i] : (ushort_t)0;
                }
                *(v16bf_a16*)la = *(v16bf_a16*)tu;
            }
        }
        // ---- B tile (always in-bounds: Bt is zero padded) ----
        ushort_t* lb = &lB[buf][ldRow * LDSS + ldKh];
        const ushort_t* B = Bg + k0;
#ifdef ASYNC_LDS
        __builtin_amdgcn_global_load_async_to_lds_b128(
            (v4i_g*)(void*)B, (v4i_l*)(void*)lb, 0, 0);
        __builtin_amdgcn_global_load_async_to_lds_b128(
            (v4i_g*)(void*)B, (v4i_l*)(void*)lb, 16, 0);
#else
        *(v16bf_a16*)lb = *(const v16bf_a16*)B;
#endif
    };

    v8f acc[2][4] = {};
    const int lrow = lane & 15;
    const int lk2  = (lane >> 4) * 16;

    int cur = 0;
    stage(0, 0);
    for (int k0 = 0; k0 < Kpad; k0 += BK) {
        wait_async_zero();                     // our async copies into buf[cur] done
        __syncthreads();                       // buf[cur] staged, buf[cur^1] free
        if (k0 + BK < Kpad) stage(k0 + BK, cur ^ 1);

        v16bf fa[2], fb[4];
        #pragma unroll
        for (int i = 0; i < 2; ++i)
            fa[i] = *(const v16bf_a16*)&lA[cur][(wr + i * 16 + lrow) * LDSS + lk2];
        #pragma unroll
        for (int j = 0; j < 4; ++j)
            fb[j] = *(const v16bf_a16*)&lB[cur][(wc + j * 16 + lrow) * LDSS + lk2];

        #pragma unroll
        for (int i = 0; i < 2; ++i)
            #pragma unroll
            for (int j = 0; j < 4; ++j)
                acc[i][j] = __builtin_amdgcn_wmma_f32_16x16x32_bf16(
                    false, fa[i], false, fb[j], (short)0, acc[i][j], false, false);
        cur ^= 1;
    }

    // ---- epilogue ----
    // C layout: vgpr v, lane<16 -> (M=v, N=lane); lane>=16 -> (M=v+8, N=lane-16)
    const int lhi = lane >> 4;
    #pragma unroll
    for (int i = 0; i < 2; ++i) {
        #pragma unroll
        for (int j = 0; j < 4; ++j) {
            const int c = colBase + wc + j * 16 + lrow;
            #pragma unroll
            for (int v = 0; v < 8; ++v) {
                const int r = rowBase + wr + i * 16 + lhi * 8 + v;
                if (r < Nrows && c < Ncols) {
                    float val = acc[i][j][v];
                    if (MODE == 0) {
                        float al = alpha[m * ab_stride + c];
                        float be = beta[m * ab_stride + c];
                        ((ushort_t*)Out)[(long long)m * strideOm + (long long)r * ldc + c]
                            = f2bf(val * al + be);
                    } else {
                        float be = beta[m * ab_stride + c];
                        atomicAdd(&((float*)Out)[(long long)r * ldc + c],
                                  0.5f * (val + be));
                    }
                }
            }
        }
    }
}

// ---------------------------------------------------------------------------
// Prep kernels
// ---------------------------------------------------------------------------
__global__ void transpose_bf16(const float* __restrict__ W, ushort_t* __restrict__ Wt,
                               int K, int C, int Kpad, int Cpad)
{
    int m = blockIdx.z;
    int idx = blockIdx.x * blockDim.x + threadIdx.x;
    int tot = Kpad * Cpad;
    if (idx >= tot) return;
    int c = idx / Kpad, k = idx % Kpad;
    float v = (k < K && c < C) ? W[(long long)m * K * C + (long long)k * C + c] : 0.f;
    Wt[(long long)m * tot + idx] = f2bf(v);
}

__global__ void bn_params(const float* __restrict__ b,  const float* __restrict__ g,
                          const float* __restrict__ be, const float* __restrict__ mu,
                          const float* __restrict__ var,
                          float* __restrict__ alpha, float* __restrict__ beta,
                          int C, int Cpad)
{
    int m = blockIdx.y;
    int c = blockIdx.x * blockDim.x + threadIdx.x;
    if (c >= Cpad) return;
    float a = 0.f, bb = 0.f;
    if (c < C) {
        float s = g[m * C + c] * rsqrtf(var[m * C + c] + 1e-5f);
        a  = s;
        bb = (b[m * C + c] - mu[m * C + c]) * s + be[m * C + c];
    }
    alpha[m * Cpad + c] = a;
    beta[m * Cpad + c]  = bb;
}

__global__ void fill_f32(float* __restrict__ p, long long n, float v)
{
    long long i = (long long)blockIdx.x * blockDim.x + threadIdx.x;
    if (i < n) p[i] = v;
}

// ---------------------------------------------------------------------------
// Graph kernels
// ---------------------------------------------------------------------------
__global__ void deg_count(const int* __restrict__ src, const int* __restrict__ dst,
                          float* __restrict__ dout, float* __restrict__ din, int E)
{
    int e = blockIdx.x * blockDim.x + threadIdx.x;
    if (e >= E) return;
    atomicAdd(&dout[src[e]], 1.f);
    atomicAdd(&din[dst[e]], 1.f);
}

__global__ void rsqrt_clamp(float* __restrict__ p, int n)
{
    int i = blockIdx.x * blockDim.x + threadIdx.x;
    if (i < n) p[i] = rsqrtf(fmaxf(p[i], 1.f));
}

// xw = (x * cs) @ W ; W (Din x Dout) staged in LDS (max 16 KB)
__global__ void xw_kernel(const float* __restrict__ x, const float* __restrict__ cs,
                          const float* __restrict__ W, float* __restrict__ xw,
                          int N, int Din, int Dout)
{
    __shared__ float sW[64 * 64];
    int tid = threadIdx.x;
    for (int i = tid; i < Din * Dout; i += blockDim.x) sW[i] = W[i];
    __syncthreads();
    long long idx = (long long)blockIdx.x * blockDim.x + tid;
    if (idx >= (long long)N * Dout) return;
    int n = (int)(idx / Dout), c = (int)(idx % Dout);
    float s = cs[n];
    float acc = 0.f;
    for (int k = 0; k < Din; ++k)
        acc += x[(long long)n * Din + k] * sW[k * Dout + c];
    xw[idx] = acc * s;
}

// agg[dst[e]] += xw[src[e]]  (float4 chunks, f32 atomics; agg is L2-resident)
__global__ void scatter_kernel(const float* __restrict__ xw, const int* __restrict__ src,
                               const int* __restrict__ dst, float* __restrict__ agg,
                               int E, int D)
{
    int chunks = D >> 2;
    long long idx = (long long)blockIdx.x * blockDim.x + threadIdx.x;
    long long tot = (long long)E * chunks;
    if (idx >= tot) return;
    int e = (int)(idx / chunks);
    int f = (int)(idx % chunks) * 4;
    int s = src[e], d = dst[e];
    float4 v = *(const float4*)&xw[(long long)s * D + f];
    atomicAdd(&agg[(long long)d * D + f + 0], v.x);
    atomicAdd(&agg[(long long)d * D + f + 1], v.y);
    atomicAdd(&agg[(long long)d * D + f + 2], v.z);
    atomicAdd(&agg[(long long)d * D + f + 3], v.w);
}

__global__ void finalize_kernel(const float* __restrict__ agg, const float* __restrict__ cd,
                                const float* __restrict__ b, float* __restrict__ out,
                                int N, int D, int relu)
{
    long long idx = (long long)blockIdx.x * blockDim.x + threadIdx.x;
    if (idx >= (long long)N * D) return;
    int n = (int)(idx / D), c = (int)(idx % D);
    float v = agg[idx] * cd[n] + b[c];
    out[idx] = relu ? fmaxf(v, 0.f) : v;
}

// ---------------------------------------------------------------------------
static inline int cdiv(long long a, long long b) { return (int)((a + b - 1) / b); }

extern "C" void kernel_launch(void* const* d_in, const int* in_sizes, int n_in,
                              void* d_out, int out_size, void* d_ws, size_t ws_size,
                              hipStream_t stream)
{
    const float* h   = (const float*)d_in[0];
    const int*   src = (const int*)  d_in[1];
    const int*   dst = (const int*)  d_in[2];
    const float* W1  = (const float*)d_in[3];
    const float* b1  = (const float*)d_in[4];
    const float* g1  = (const float*)d_in[5];
    const float* be1 = (const float*)d_in[6];
    const float* m1  = (const float*)d_in[7];
    const float* v1  = (const float*)d_in[8];
    const float* W2  = (const float*)d_in[9];
    const float* b2  = (const float*)d_in[10];
    const float* g2  = (const float*)d_in[11];
    const float* be2 = (const float*)d_in[12];
    const float* m2  = (const float*)d_in[13];
    const float* v2  = (const float*)d_in[14];
    const float* Wd  = (const float*)d_in[15];
    const float* bd  = (const float*)d_in[16];
    const float* Wg0 = (const float*)d_in[17];
    const float* bg0 = (const float*)d_in[18];
    const float* Wg1 = (const float*)d_in[19];
    const float* bg1 = (const float*)d_in[20];
    const float* Wg2 = (const float*)d_in[21];
    const float* bg2 = (const float*)d_in[22];

    const int DIN = 2000, H1 = 500, H2 = 128, HD = 64;
    const int K1P = 2048, C1P = 512, HDP = 128;   // padded dims
    const int E = in_sizes[1];
    const int N = in_sizes[0] / (2 * DIN);

    // ---- workspace carve-out ----
    size_t off = 0;
    auto alloc = [&](size_t bytes) -> void* {
        void* p = (char*)d_ws + off;
        off += (bytes + 255) & ~(size_t)255;
        return p;
    };
    ushort_t* W1t = (ushort_t*)alloc((size_t)2 * C1P * K1P * 2);
    ushort_t* W2t = (ushort_t*)alloc((size_t)2 * H2 * C1P * 2);
    ushort_t* Wdt = (ushort_t*)alloc((size_t)2 * HDP * H2 * 2);  // cols padded 64->128
    float* alpha1 = (float*)alloc((size_t)2 * C1P * 4);
    float* beta1  = (float*)alloc((size_t)2 * C1P * 4);
    float* alpha2 = (float*)alloc((size_t)2 * H2 * 4);
    float* beta2  = (float*)alloc((size_t)2 * H2 * 4);
    ushort_t* z1  = (ushort_t*)alloc((size_t)2 * N * C1P * 2);
    ushort_t* z2  = (ushort_t*)alloc((size_t)2 * N * H2 * 2);
    float* feat   = (float*)alloc((size_t)N * HD * 4);
    float* cs     = (float*)alloc((size_t)N * 4);
    float* cd     = (float*)alloc((size_t)N * 4);
    float* xw     = (float*)alloc((size_t)N * 64 * 4);
    float* agg    = (float*)alloc((size_t)N * 64 * 4);
    float* x1     = (float*)alloc((size_t)N * 64 * 4);
    float* x2     = (float*)alloc((size_t)N * 32 * 4);

    // ---- weight prep (bf16 transpose + BN fold) ----
    transpose_bf16<<<dim3(cdiv((long long)C1P * K1P, 256), 1, 2), 256, 0, stream>>>(
        W1, W1t, DIN, H1, K1P, C1P);
    transpose_bf16<<<dim3(cdiv((long long)H2 * C1P, 256), 1, 2), 256, 0, stream>>>(
        W2, W2t, H1, H2, C1P, H2);
    transpose_bf16<<<dim3(cdiv((long long)HDP * H2, 256), 1, 2), 256, 0, stream>>>(
        Wd, Wdt, H2, HD, H2, HDP);
    bn_params<<<dim3(cdiv(C1P, 256), 2), 256, 0, stream>>>(b1, g1, be1, m1, v1,
                                                           alpha1, beta1, H1, C1P);
    bn_params<<<dim3(cdiv(H2, 256), 2), 256, 0, stream>>>(b2, g2, be2, m2, v2,
                                                          alpha2, beta2, H2, H2);
    fill_f32<<<cdiv((long long)N * HD, 256), 256, 0, stream>>>(feat, (long long)N * HD, 0.f);

    const int rb = cdiv(N, BM);
    // layer 1: h (fp32, K=2000) @ W1t -> z1 bf16 [2,N,512], BN epilogue
    gemm_wmma<0, true><<<dim3(rb, C1P / BN, 2), 256, 0, stream>>>(
        h, (long long)N * DIN, DIN, DIN, K1P,
        W1t, (long long)C1P * K1P, K1P,
        z1, (long long)N * C1P, C1P, alpha1, beta1, C1P, N, C1P);
    // layer 2: z1 (bf16, K=512) @ W2t -> z2 bf16 [2,N,128], BN epilogue
    gemm_wmma<0, false><<<dim3(rb, H2 / BN, 2), 256, 0, stream>>>(
        z1, (long long)N * C1P, C1P, C1P, C1P,
        W2t, (long long)H2 * C1P, C1P,
        z2, (long long)N * H2, H2, alpha2, beta2, H2, N, H2);
    // decoder: z2 @ Wdt -> feat fp32 [N,64], mean over modalities (0.5*atomicAdd)
    gemm_wmma<1, false><<<dim3(rb, HDP / BN, 2), 256, 0, stream>>>(
        z2, (long long)N * H2, H2, H2, H2,
        Wdt, (long long)HDP * H2, H2,
        feat, 0, HD, nullptr, bd, HD, N, HD);

    // ---- graph normalization ----
    fill_f32<<<cdiv(N, 256), 256, 0, stream>>>(cs, N, 0.f);
    fill_f32<<<cdiv(N, 256), 256, 0, stream>>>(cd, N, 0.f);
    deg_count<<<cdiv(E, 256), 256, 0, stream>>>(src, dst, cs, cd, E);
    rsqrt_clamp<<<cdiv(N, 256), 256, 0, stream>>>(cs, N);
    rsqrt_clamp<<<cdiv(N, 256), 256, 0, stream>>>(cd, N);

    // ---- GraphConv layer 0: 64 -> 64, ReLU ----
    fill_f32<<<cdiv((long long)N * 64, 256), 256, 0, stream>>>(agg, (long long)N * 64, 0.f);
    xw_kernel<<<cdiv((long long)N * 64, 256), 256, 0, stream>>>(feat, cs, Wg0, xw, N, 64, 64);
    scatter_kernel<<<cdiv((long long)E * 16, 256), 256, 0, stream>>>(xw, src, dst, agg, E, 64);
    finalize_kernel<<<cdiv((long long)N * 64, 256), 256, 0, stream>>>(agg, cd, bg0, x1, N, 64, 1);

    // ---- GraphConv layer 1: 64 -> 32, ReLU ----
    fill_f32<<<cdiv((long long)N * 32, 256), 256, 0, stream>>>(agg, (long long)N * 32, 0.f);
    xw_kernel<<<cdiv((long long)N * 32, 256), 256, 0, stream>>>(x1, cs, Wg1, xw, N, 64, 32);
    scatter_kernel<<<cdiv((long long)E * 8, 256), 256, 0, stream>>>(xw, src, dst, agg, E, 32);
    finalize_kernel<<<cdiv((long long)N * 32, 256), 256, 0, stream>>>(agg, cd, bg1, x2, N, 32, 1);

    // ---- GraphConv layer 2: 32 -> 4, linear -> d_out ----
    fill_f32<<<cdiv((long long)N * 4, 256), 256, 0, stream>>>(agg, (long long)N * 4, 0.f);
    xw_kernel<<<cdiv((long long)N * 4, 256), 256, 0, stream>>>(x2, cs, Wg2, xw, N, 32, 4);
    scatter_kernel<<<cdiv((long long)E * 1, 256), 256, 0, stream>>>(xw, src, dst, agg, E, 4);
    finalize_kernel<<<cdiv((long long)N * 4, 256), 256, 0, stream>>>(agg, cd, bg2,
                                                                     (float*)d_out, N, 4, 0);
    (void)n_in; (void)out_size; (void)ws_size;
}
